// DRAW_32100585570356
// MI455X (gfx1250) — matrix operand order repaired and checked by reading, not compiled, not run on hardware
//
#include <hip/hip_runtime.h>
#include <hip/hip_bf16.h>

// ---------------- problem constants ----------------
#define BB     512
#define HH     128
#define WWIDTH 128
#define RSZ    12            // RS == WS == 12
#define LL     128
#define HID    512
#define TT     16
#define ENC_IN 800           // 2*12*12 + 512
#define G4     2048          // 4*HID
#define NPIX   (HH*WWIDTH)   // 16384
#define NWRIT_PAD 192        // 144 padded up to 3*64

typedef __attribute__((ext_vector_type(16))) _Float16 v16h;
typedef __attribute__((ext_vector_type(8)))  _Float16 v8h;
typedef __attribute__((ext_vector_type(8)))  float    v8f;

// =====================================================================
// WMMA GEMM:  C[M,N] (f32) = A[M,K](f16) * Wt[N,K]^T (f16) (+bias0+bias1) (+=C)
// Block = 128 threads = 4 waves. Each wave computes a 32(M) x 64(N) strip:
// 8 v_wmma_f32_16x16x32_f16 accumulators; 2 A-fragments x 4 B-fragments per
// k-step (B reused across both M-tiles, A reused across all 4 N-tiles).
// All 6 fragment loads are issued as one clause before the 8 WMMAs so the
// compiler can stage s_wait_loadcnt instead of stalling per fragment.
// grid = (N/64, M/128). M mult of 128, N mult of 64, K mult of 32.
// =====================================================================
__global__ __launch_bounds__(128)
void gemm_f16_wmma(const _Float16* __restrict__ A, const _Float16* __restrict__ Wt,
                   const float* __restrict__ bias0, const float* __restrict__ bias1,
                   float* __restrict__ C, int M, int N, int K, int accumulate)
{
    const int wave = threadIdx.x >> 5;
    const int lane = threadIdx.x & 31;
    const int hl   = lane >> 4;     // half-wave: 0 or 1
    const int l16  = lane & 15;
    const int m0   = blockIdx.y * 128 + wave * 32;   // wave owns rows m0..m0+31
    const int n0   = blockIdx.x * 64;

    v8f acc[2][4] = {};

    // A fragment rows for the two M-tiles
    const _Float16* arow0 = A + (size_t)(m0 + l16) * K + hl * 8;
    const _Float16* arow1 = arow0 + (size_t)16 * K;

    for (int kk = 0; kk < K; kk += 32) {
        // ---- issue all loads first (one clause), then compute ----
        // A fragments: elems 0..7 -> K = kk + hl*8 + e ; elems 8..15 -> +16
        v8h a0lo = *(const v8h*)(arow0 + kk);
        v8h a0hi = *(const v8h*)(arow0 + kk + 16);
        v8h a1lo = *(const v8h*)(arow1 + kk);
        v8h a1hi = *(const v8h*)(arow1 + kk + 16);
        // B fragments: col n = n0+t*16+l16 ; elems 0..7 -> K = kk + hl*8 + e ; elems 8..15 -> +16
        v16h bfrag[4];
#pragma unroll
        for (int t = 0; t < 4; ++t) {
            const _Float16* bp = Wt + (size_t)(n0 + t * 16 + l16) * K + kk + hl * 8;
            bfrag[t] = __builtin_shufflevector(*(const v8h*)bp, *(const v8h*)(bp + 16),
                           0,1,2,3,4,5,6,7,8,9,10,11,12,13,14,15);
        }

        // prefetch next K-slab of weights (global_prefetch_b8 on gfx1250)
        if (kk + 32 < K)
            __builtin_prefetch(Wt + (size_t)(n0 + l16) * K + kk + 32, 0, 1);

        v16h a0 = __builtin_shufflevector(a0lo, a0hi,
                      0,1,2,3,4,5,6,7,8,9,10,11,12,13,14,15);
        v16h a1 = __builtin_shufflevector(a1lo, a1hi,
                      0,1,2,3,4,5,6,7,8,9,10,11,12,13,14,15);

#pragma unroll
        for (int t = 0; t < 4; ++t) {
            acc[0][t] = __builtin_amdgcn_wmma_f32_16x16x32_f16(
                            false, a0, false, bfrag[t], (short)0, acc[0][t], false, false);
            acc[1][t] = __builtin_amdgcn_wmma_f32_16x16x32_f16(
                            false, a1, false, bfrag[t], (short)0, acc[1][t], false, false);
        }
    }

#pragma unroll
    for (int mt = 0; mt < 2; ++mt) {
#pragma unroll
        for (int t = 0; t < 4; ++t) {
            int n = n0 + t * 16 + l16;
            float bv = 0.f;
            if (bias0) bv += bias0[n];
            if (bias1) bv += bias1[n];
#pragma unroll
            for (int r = 0; r < 8; ++r) {
                int m = m0 + mt * 16 + r + hl * 8;   // C/D layout: elem r, halves split M by 8
                size_t idx = (size_t)m * N + n;
                float v = acc[mt][t][r] + bv;
                if (accumulate) v += C[idx];
                C[idx] = v;
            }
        }
    }
}

// =====================================================================
// Attention window: res = h_dec @ Wattn^T + battn (5 dots of 512),
// then filterbank Fx(12x128), Fy(12x128) row-normalized, inten.
// One 256-thread block per batch sample.
// =====================================================================
__global__ __launch_bounds__(256)
void attn_kernel(const float* __restrict__ h, const float* __restrict__ Wattn,
                 const float* __restrict__ battn,
                 float* __restrict__ Fx, float* __restrict__ Fy,
                 float* __restrict__ inten)
{
    __shared__ float red[256];
    __shared__ float res[5];
    __shared__ float prm[4];
    __shared__ float F[2 * RSZ * 128];     // raw Fx rows 0..11, Fy rows 12..23
    __shared__ float rs[24];

    const int b = blockIdx.x, tid = threadIdx.x;
    const float* hb = h + (size_t)b * HID;

    for (int r = 0; r < 5; ++r) {
        red[tid] = hb[tid] * Wattn[r * HID + tid] + hb[tid + 256] * Wattn[r * HID + tid + 256];
        __syncthreads();
        for (int s = 128; s > 0; s >>= 1) {
            if (tid < s) red[tid] += red[tid + s];
            __syncthreads();
        }
        if (tid == 0) res[r] = red[0] + battn[r];
        __syncthreads();
    }
    if (tid == 0) {
        prm[0] = (WWIDTH + 1) * (res[0] + 1.f) * 0.5f;                 // gx
        prm[1] = (HH + 1) * (res[1] + 1.f) * 0.5f;                     // gy
        prm[2] = (127.f) * __expf(res[3]) / (RSZ - 1);                 // stride
        prm[3] = __expf(res[2]);                                       // var
        inten[b] = __expf(res[4]);
    }
    __syncthreads();
    const float gx = prm[0], gy = prm[1], stride = prm[2];
    const float inv2v = 1.f / (2.f * prm[3]);

    for (int q = 0; q < 12; ++q) {              // 3072 raw filter values
        int p = tid + q * 256;
        int which = p / 1536;                   // 0=Fx, 1=Fy
        int rem = p - which * 1536;
        int i = rem >> 7, a = rem & 127;
        float mu = (which ? gy : gx) + ((float)i + 0.5f - 6.f) * stride;
        float d = (float)a - mu;
        F[p] = __expf(-d * d * inv2v);
    }
    __syncthreads();
    if (tid < 24) {
        float s = 0.f;
        for (int a = 0; a < 128; ++a) s += F[tid * 128 + a];
        rs[tid] = 1.f / fmaxf(s, 1e-8f);
    }
    __syncthreads();
    for (int q = 0; q < 12; ++q) {
        int p = tid + q * 256;
        float v = F[p] * rs[p >> 7];
        if (p < 1536) Fx[(size_t)b * 1536 + p]          = v;
        else          Fy[(size_t)b * 1536 + (p - 1536)] = v;
    }
}

// =====================================================================
// Fused read: x_error = x - sigmoid(canvas) on the fly; single pass over
// x/canvas per step. glimpse = inten * [Fy@x@Fx^T ; Fy@xe@Fx^T] -> encin f16,
// plus encin[288:800] = f16(h_dec). One block per batch sample.
// =====================================================================
__global__ __launch_bounds__(256)
void glimpse_kernel(const float* __restrict__ x, const float* __restrict__ canvas,
                    const float* __restrict__ Fx, const float* __restrict__ Fy,
                    const float* __restrict__ inten, const float* __restrict__ h_dec,
                    _Float16* __restrict__ encin)
{
    __shared__ float sFx[1536], sFy[1536], t_x[1536], t_e[1536];
    const int b = blockIdx.x, tid = threadIdx.x;
    for (int q = tid; q < 1536; q += 256) {
        sFx[q] = Fx[(size_t)b * 1536 + q];
        sFy[q] = Fy[(size_t)b * 1536 + q];
    }
    __syncthreads();
    const float* xb = x + (size_t)b * NPIX;
    const float* cb = canvas + (size_t)b * NPIX;
    for (int q = 0; q < 6; ++q) {                 // t[h][j] = (x @ Fx^T)[h][j]
        int p = tid + q * 256;
        int hrow = p / 12, j = p - hrow * 12;
        const float* xr = xb + hrow * 128;
        const float* cr = cb + hrow * 128;
        float sx = 0.f, se = 0.f;
        for (int w = 0; w < 128; ++w) {
            float fx = sFx[j * 128 + w];
            float xv = xr[w];
            float xe = xv - 1.f / (1.f + __expf(-cr[w]));
            sx += xv * fx;
            se += xe * fx;
        }
        t_x[p] = sx; t_e[p] = se;
    }
    __syncthreads();
    const float I = inten[b];
    if (tid < 144) {                              // g[i][j] = (Fy @ t)[i][j]
        int i = tid / 12, j = tid - i * 12;
        float gxv = 0.f, gev = 0.f;
        for (int hh = 0; hh < 128; ++hh) {
            float fy = sFy[i * 128 + hh];
            gxv += fy * t_x[hh * 12 + j];
            gev += fy * t_e[hh * 12 + j];
        }
        encin[(size_t)b * ENC_IN + tid]       = (_Float16)(I * gxv);
        encin[(size_t)b * ENC_IN + 144 + tid] = (_Float16)(I * gev);
    }
    for (int k = tid; k < HID; k += 256)
        encin[(size_t)b * ENC_IN + 288 + k] = (_Float16)h_dec[(size_t)b * HID + k];
}

// ---------------- LSTM pointwise update ----------------
__global__ void lstm_update(const float* __restrict__ gates, float* __restrict__ h,
                            float* __restrict__ c, _Float16* __restrict__ h_h)
{
    long i = (long)blockIdx.x * blockDim.x + threadIdx.x;
    if (i >= (long)BB * HID) return;
    int b = (int)(i >> 9), k = (int)(i & 511);
    const float* g = gates + (size_t)b * G4;
    float si = 1.f / (1.f + __expf(-g[k]));
    float sf = 1.f / (1.f + __expf(-g[512 + k]));
    float so = 1.f / (1.f + __expf(-g[1536 + k]));
    float tg = tanhf(g[1024 + k]);
    float cn = sf * c[i] + si * tg;
    float hn = so * tanhf(cn);
    c[i] = cn; h[i] = hn; h_h[i] = (_Float16)hn;
}

// ---------------- reparameterization sample + mu/logvar output ----------------
__global__ void sample_kernel(const float* __restrict__ s, const float* __restrict__ eps_t,
                              float* __restrict__ out_mu, float* __restrict__ out_lv,
                              _Float16* __restrict__ z_h, int t)
{
    long i = (long)blockIdx.x * blockDim.x + threadIdx.x;
    if (i >= (long)BB * LL) return;
    int b = (int)(i >> 7), l = (int)(i & 127);
    float mu = s[(size_t)b * 256 + l];
    float lv = s[(size_t)b * 256 + 128 + l];
    float z  = mu + __expf(0.5f * lv) * eps_t[i];
    size_t oidx = ((size_t)b * TT + t) * LL + l;   // (B,T,L) after swapaxes
    out_mu[oidx] = mu;
    out_lv[oidx] = lv;
    z_h[i] = (_Float16)z;
}

// ---------------- write: canvas += Fy^T @ w @ Fx / inten ----------------
__global__ __launch_bounds__(256)
void write_canvas(const float* __restrict__ wpatch /* B x 192, first 144 used */,
                  const float* __restrict__ Fx, const float* __restrict__ Fy,
                  const float* __restrict__ inten, float* __restrict__ canvas)
{
    __shared__ float sFx[1536], sFy[1536], sw[144], u[1536];
    const int b = blockIdx.x, tid = threadIdx.x;
    for (int q = tid; q < 1536; q += 256) {
        sFx[q] = Fx[(size_t)b * 1536 + q];
        sFy[q] = Fy[(size_t)b * 1536 + q];
    }
    if (tid < 144) sw[tid] = wpatch[(size_t)b * NWRIT_PAD + tid];
    __syncthreads();
    for (int q = 0; q < 6; ++q) {                 // u[h][j] = sum_i Fy[i][h]*w[i][j]
        int p = tid + q * 256;
        int hrow = p / 12, j = p - hrow * 12;
        float s = 0.f;
        for (int i = 0; i < 12; ++i) s += sFy[i * 128 + hrow] * sw[i * 12 + j];
        u[p] = s;
    }
    __syncthreads();
    const float invI = 1.f / inten[b];
    float* cb = canvas + (size_t)b * NPIX;
    for (int q = 0; q < 64; ++q) {                // canvas[h][a] += u[h][:].Fx[:][a]/I
        int p = tid + q * 256;
        int hrow = p >> 7, a = p & 127;
        float s = 0.f;
        for (int j = 0; j < 12; ++j) s += u[hrow * 12 + j] * sFx[j * 128 + a];
        cb[p] += s * invI;
    }
}

// ---------------- init / conversion helpers ----------------
__global__ void f32_to_f16_k(const float* __restrict__ s, _Float16* __restrict__ d, long n) {
    long i = (long)blockIdx.x * blockDim.x + threadIdx.x;
    if (i < n) d[i] = (_Float16)s[i];
}
__global__ void zero_f32_k(float* p, long n) {
    long i = (long)blockIdx.x * blockDim.x + threadIdx.x;
    if (i < n) p[i] = 0.f;
}
__global__ void bcast_f32_k(const float* __restrict__ s, float* __restrict__ d, int per, long n) {
    long i = (long)blockIdx.x * blockDim.x + threadIdx.x;
    if (i < n) d[i] = s[i % per];
}
__global__ void bcast_state_k(const float* __restrict__ s, float* __restrict__ d,
                              _Float16* __restrict__ dh, int per, long n) {
    long i = (long)blockIdx.x * blockDim.x + threadIdx.x;
    if (i < n) { float v = s[i % per]; d[i] = v; dh[i] = (_Float16)v; }
}
__global__ void pad_writer_k(const float* __restrict__ Wsrc /*144x512*/,
                             const float* __restrict__ bsrc /*144*/,
                             _Float16* __restrict__ Wdst /*192x512*/,
                             float* __restrict__ bdst /*192*/) {
    long i = (long)blockIdx.x * blockDim.x + threadIdx.x;
    if (i < (long)NWRIT_PAD * HID) {
        int r = (int)(i / HID);
        Wdst[i] = (r < 144) ? (_Float16)Wsrc[i] : (_Float16)0.f;
    }
    if (i < NWRIT_PAD) bdst[i] = (i < 144) ? bsrc[i] : 0.f;
}
__global__ void sigmoid_inplace_k(float* p, long n) {
    long i = (long)blockIdx.x * blockDim.x + threadIdx.x;
    if (i < n) p[i] = 1.f / (1.f + __expf(-p[i]));
}

// =====================================================================
// Host driver
// =====================================================================
extern "C" void kernel_launch(void* const* d_in, const int* in_sizes, int n_in,
                              void* d_out, int out_size, void* d_ws, size_t ws_size,
                              hipStream_t stream)
{
    const float* x          = (const float*)d_in[0];
    const float* eps        = (const float*)d_in[1];
    const float* canvas_in  = (const float*)d_in[3];
    const float* h_dec_init = (const float*)d_in[4];
    const float* h_enc_init = (const float*)d_in[5];
    const float* W_ih_enc   = (const float*)d_in[6];
    const float* b_ih_enc   = (const float*)d_in[7];
    const float* W_hh_enc   = (const float*)d_in[8];
    const float* b_hh_enc   = (const float*)d_in[9];
    const float* W_ih_dec   = (const float*)d_in[10];
    const float* b_ih_dec   = (const float*)d_in[11];
    const float* W_hh_dec   = (const float*)d_in[12];
    const float* b_hh_dec   = (const float*)d_in[13];
    const float* W_samp     = (const float*)d_in[14];
    const float* b_samp     = (const float*)d_in[15];
    const float* W_rattn    = (const float*)d_in[16];
    const float* b_rattn    = (const float*)d_in[17];
    const float* W_wattn    = (const float*)d_in[18];
    const float* b_wattn    = (const float*)d_in[19];
    const float* W_writer   = (const float*)d_in[20];
    const float* b_writer   = (const float*)d_in[21];

    float* out     = (float*)d_out;
    float* canvas  = out;                                  // raw canvas lives in d_out
    float* out_mu  = out + (size_t)BB * NPIX;
    float* out_lv  = out_mu + (size_t)BB * TT * LL;

    // ---- workspace carve-out (256B aligned) ----
    size_t o = 0;
    auto take = [&](size_t bytes) -> void* {
        void* p = (char*)d_ws + o;
        o += (bytes + 255) & ~(size_t)255;
        return p;
    };
    _Float16* WihEnc_h = (_Float16*)take((size_t)G4 * ENC_IN * 2);
    _Float16* WhhEnc_h = (_Float16*)take((size_t)G4 * HID * 2);
    _Float16* WihDec_h = (_Float16*)take((size_t)G4 * LL * 2);
    _Float16* WhhDec_h = (_Float16*)take((size_t)G4 * HID * 2);
    _Float16* Wsamp_h  = (_Float16*)take((size_t)256 * HID * 2);
    _Float16* Wwrit_h  = (_Float16*)take((size_t)NWRIT_PAD * HID * 2);
    float*    bwrit_p  = (float*)take(NWRIT_PAD * 4);
    float*    h_enc    = (float*)take((size_t)BB * HID * 4);
    float*    c_enc    = (float*)take((size_t)BB * HID * 4);
    float*    h_dec    = (float*)take((size_t)BB * HID * 4);
    float*    c_dec    = (float*)take((size_t)BB * HID * 4);
    _Float16* h_enc_h  = (_Float16*)take((size_t)BB * HID * 2);
    _Float16* h_dec_h  = (_Float16*)take((size_t)BB * HID * 2);
    _Float16* encin_h  = (_Float16*)take((size_t)BB * ENC_IN * 2);
    _Float16* z_h      = (_Float16*)take((size_t)BB * LL * 2);
    float*    gates    = (float*)take((size_t)BB * G4 * 4);   // reused for s / wpatch
    float*    Fx       = (float*)take((size_t)BB * 1536 * 4);
    float*    Fy       = (float*)take((size_t)BB * 1536 * 4);
    float*    inten    = (float*)take((size_t)BB * 4);
    (void)ws_size; (void)in_sizes; (void)n_in; (void)out_size;

    auto nb = [](long n) { return (int)((n + 255) / 256); };

    // ---- one-time (per launch) weight conversion + state init ----
    f32_to_f16_k<<<nb((long)G4 * ENC_IN), 256, 0, stream>>>(W_ih_enc, WihEnc_h, (long)G4 * ENC_IN);
    f32_to_f16_k<<<nb((long)G4 * HID),    256, 0, stream>>>(W_hh_enc, WhhEnc_h, (long)G4 * HID);
    f32_to_f16_k<<<nb((long)G4 * LL),     256, 0, stream>>>(W_ih_dec, WihDec_h, (long)G4 * LL);
    f32_to_f16_k<<<nb((long)G4 * HID),    256, 0, stream>>>(W_hh_dec, WhhDec_h, (long)G4 * HID);
    f32_to_f16_k<<<nb((long)256 * HID),   256, 0, stream>>>(W_samp,   Wsamp_h,  (long)256 * HID);
    pad_writer_k<<<nb((long)NWRIT_PAD * HID), 256, 0, stream>>>(W_writer, b_writer, Wwrit_h, bwrit_p);

    bcast_f32_k<<<nb((long)BB * NPIX), 256, 0, stream>>>(canvas_in, canvas, NPIX, (long)BB * NPIX);
    bcast_state_k<<<nb((long)BB * HID), 256, 0, stream>>>(h_enc_init, h_enc, h_enc_h, HID, (long)BB * HID);
    bcast_state_k<<<nb((long)BB * HID), 256, 0, stream>>>(h_dec_init, h_dec, h_dec_h, HID, (long)BB * HID);
    zero_f32_k<<<nb((long)BB * HID), 256, 0, stream>>>(c_enc, (long)BB * HID);
    zero_f32_k<<<nb((long)BB * HID), 256, 0, stream>>>(c_dec, (long)BB * HID);

    const dim3 gGate(G4 / 64, BB / 128);        // 32 x 4
    const dim3 gSamp(256 / 64, BB / 128);       // 4 x 4
    const dim3 gWrit(NWRIT_PAD / 64, BB / 128); // 3 x 4

    for (int t = 0; t < TT; ++t) {
        // read attention + glimpse (uses h_dec(t-1), canvas(t-1))
        attn_kernel<<<BB, 256, 0, stream>>>(h_dec, W_rattn, b_rattn, Fx, Fy, inten);
        glimpse_kernel<<<BB, 256, 0, stream>>>(x, canvas, Fx, Fy, inten, h_dec, encin_h);

        // encoder LSTM
        gemm_f16_wmma<<<gGate, 128, 0, stream>>>(encin_h, WihEnc_h, b_ih_enc, b_hh_enc,
                                                 gates, BB, G4, ENC_IN, 0);
        gemm_f16_wmma<<<gGate, 128, 0, stream>>>(h_enc_h, WhhEnc_h, nullptr, nullptr,
                                                 gates, BB, G4, HID, 1);
        lstm_update<<<nb((long)BB * HID), 256, 0, stream>>>(gates, h_enc, c_enc, h_enc_h);

        // sampler + reparameterization
        gemm_f16_wmma<<<gSamp, 128, 0, stream>>>(h_enc_h, Wsamp_h, b_samp, nullptr,
                                                 gates, BB, 256, HID, 0);
        sample_kernel<<<nb((long)BB * LL), 256, 0, stream>>>(gates, eps + (size_t)t * BB * LL,
                                                             out_mu, out_lv, z_h, t);

        // decoder LSTM (uses h_dec(t-1))
        gemm_f16_wmma<<<gGate, 128, 0, stream>>>(z_h, WihDec_h, b_ih_dec, b_hh_dec,
                                                 gates, BB, G4, LL, 0);
        gemm_f16_wmma<<<gGate, 128, 0, stream>>>(h_dec_h, WhhDec_h, nullptr, nullptr,
                                                 gates, BB, G4, HID, 1);
        lstm_update<<<nb((long)BB * HID), 256, 0, stream>>>(gates, h_dec, c_dec, h_dec_h);

        // writer + write attention + canvas update
        gemm_f16_wmma<<<gWrit, 128, 0, stream>>>(h_dec_h, Wwrit_h, bwrit_p, nullptr,
                                                 gates, BB, NWRIT_PAD, HID, 0);
        attn_kernel<<<BB, 256, 0, stream>>>(h_dec, W_wattn, b_wattn, Fx, Fy, inten);
        write_canvas<<<BB, 256, 0, stream>>>(gates, Fx, Fy, inten, canvas);
    }

    sigmoid_inplace_k<<<nb((long)BB * NPIX), 256, 0, stream>>>(canvas, (long)BB * NPIX);
}
